// SioConvMemoryBlock_69458211111500
// MI455X (gfx1250) — matrix operand order repaired and checked by reading, not compiled, not run on hardware
//
#include <hip/hip_runtime.h>
#include <math.h>

// ---------------------------------------------------------------------------
// Types for CDNA5 WMMA
// ---------------------------------------------------------------------------
typedef __bf16 bf16_t;
typedef bf16_t v16bf __attribute__((ext_vector_type(16)));
typedef float  v8f   __attribute__((ext_vector_type(8)));
typedef unsigned int u32x4 __attribute__((ext_vector_type(4)));

union FragBF { u32x4 q[2]; v16bf v; };

#define DIM   1024
#define FFH   4096
#define NHEAD 8
#define HDIM  128
#define CHUNK 64
#define NC    32
#define BATCH 4
#define SEQ   2048
#define MTOK  (BATCH*SEQ)          // 8192 tokens

// ---------------------------------------------------------------------------
// CDNA5 async global->LDS copy (ASYNCcnt-tracked DMA, no VGPR round trip)
// ---------------------------------------------------------------------------
__device__ __forceinline__ void async_load_b128(const void* gptr, void* lptr)
{
    unsigned lds_off = (unsigned)(unsigned long long)lptr;  // low 32 bits = LDS byte addr
    asm volatile("global_load_async_to_lds_b128 %0, %1, off"
                 :
                 : "v"(lds_off), "v"(gptr)
                 : "memory");
}

__device__ __forceinline__ void wait_async0()
{
    asm volatile("s_wait_asynccnt 0x0" ::: "memory");
}

// ---------------------------------------------------------------------------
// Weight transpose + f32 -> bf16 convert:  W[K,N] f32  ->  Wt[N,K] bf16
// ---------------------------------------------------------------------------
__global__ void transpose_to_bf16_kernel(const float* __restrict__ W,
                                         bf16_t* __restrict__ Wt,
                                         int K, int N)
{
    __shared__ float tile[32][33];
    int k0 = blockIdx.y * 32;
    int n0 = blockIdx.x * 32;
    int tx = threadIdx.x;           // 0..31
    int ty = threadIdx.y;           // 0..7
    for (int i = ty; i < 32; i += 8)
        tile[i][tx] = W[(size_t)(k0 + i) * N + (n0 + tx)];
    __syncthreads();
    for (int i = ty; i < 32; i += 8)
        Wt[(size_t)(n0 + i) * K + (k0 + tx)] = (bf16_t)tile[tx][i];
}

// ---------------------------------------------------------------------------
// RMSNorm (per row of DIM) -> bf16 output
// ---------------------------------------------------------------------------
__global__ __launch_bounds__(256) void rmsnorm_bf16_kernel(
    const float* __restrict__ x, const float* __restrict__ g,
    bf16_t* __restrict__ out)
{
    __shared__ float red[256];
    int row = blockIdx.x;
    int tid = threadIdx.x;
    const float* xr = x + (size_t)row * DIM;
    float s = 0.f;
    for (int i = tid; i < DIM; i += 256) { float v = xr[i]; s += v * v; }
    red[tid] = s;
    __syncthreads();
    for (int o = 128; o > 0; o >>= 1) {
        if (tid < o) red[tid] += red[tid + o];
        __syncthreads();
    }
    float rs = rsqrtf(red[0] * (1.0f / DIM) + 1e-6f);
    bf16_t* orow = out + (size_t)row * DIM;
    for (int i = tid; i < DIM; i += 256)
        orow[i] = (bf16_t)(xr[i] * rs * g[i]);
}

// ---------------------------------------------------------------------------
// WMMA GEMM:  C[M,N] = act( A_bf16[M,K] @ Wt_bf16[N,K]^T + bias[N] )
//   act: 0 = none, 2 = silu
//   Block tile 128x128, 8 waves (4 in M x 2 in N), wave tile 32x64.
//   Double-buffered LDS filled via global_load_async_to_lds_b128,
//   synchronized with s_wait_asynccnt + one barrier per K-step.
// ---------------------------------------------------------------------------
__global__ __launch_bounds__(256) void gemm_bf16_wmma_kernel(
    const bf16_t* __restrict__ A, const bf16_t* __restrict__ Wt,
    const float* __restrict__ bias, float* __restrict__ C,
    int Mdim, int Ndim, int Kdim, int act)
{
    __shared__ bf16_t sA[2][128 * 64];
    __shared__ bf16_t sB[2][128 * 64];

    const int bm   = blockIdx.y * 128;
    const int bn   = blockIdx.x * 128;
    const int tid  = threadIdx.x;
    const int wave = tid >> 5;
    const int lane = tid & 31;
    const int wm   = (wave & 3) * 32;   // wave offset in M
    const int wn   = (wave >> 2) * 64;  // wave offset in N
    const int lrow = lane & 15;
    const int lhalf = lane >> 4;

    v8f acc[2][4];
    v8f zero = {0.f, 0.f, 0.f, 0.f, 0.f, 0.f, 0.f, 0.f};
    for (int i = 0; i < 2; ++i)
        for (int j = 0; j < 4; ++j) acc[i][j] = zero;

    // async-issue one 128x64 A tile + 128x64 B tile into LDS buffer `buf`
    auto issue_tile = [&](int buf, int k0) {
#pragma unroll
        for (int i = 0; i < 4; ++i) {
            int seg = tid + i * 256;          // 1024 16B segments per tile
            int r = seg >> 3, c = seg & 7;
            async_load_b128(A  + (size_t)(bm + r) * Kdim + k0 + c * 8,
                            &sA[buf][r * 64 + c * 8]);
            async_load_b128(Wt + (size_t)(bn + r) * Kdim + k0 + c * 8,
                            &sB[buf][r * 64 + c * 8]);
        }
    };

    const int nk = Kdim >> 6;   // K / 64
    issue_tile(0, 0);

    for (int t = 0; t < nk; ++t) {
        wait_async0();          // own async copies into buf (t&1) are done
        __syncthreads();        // everyone's copies are done; prev reads of buf^1 done
        if (t + 1 < nk) issue_tile((t + 1) & 1, (t + 1) * 64);

        const u32x4* cA4 = (const u32x4*)sA[t & 1];
        const u32x4* cB4 = (const u32x4*)sB[t & 1];
#pragma unroll
        for (int kk = 0; kk < 2; ++kk) {
            const int kb = kk * 32;
            FragBF a[2], b[4];
#pragma unroll
            for (int mt = 0; mt < 2; ++mt) {
                int row = wm + mt * 16 + lrow;
                int ka = (kb + lhalf * 8) >> 3;       // uint4 index
                a[mt].q[0] = cA4[row * 8 + ka];
                a[mt].q[1] = cA4[row * 8 + ka + 2];   // K += 16
            }
#pragma unroll
            for (int nt = 0; nt < 4; ++nt) {
                int rowb = wn + nt * 16 + lrow;
                int kq = (kb + lhalf * 8) >> 3;
                b[nt].q[0] = cB4[rowb * 8 + kq];
                b[nt].q[1] = cB4[rowb * 8 + kq + 2];  // K += 16
            }
#pragma unroll
            for (int mt = 0; mt < 2; ++mt)
#pragma unroll
                for (int nt = 0; nt < 4; ++nt)
                    acc[mt][nt] = __builtin_amdgcn_wmma_f32_16x16x32_bf16(
                        false, a[mt].v, false, b[nt].v,
                        (short)0, acc[mt][nt], false, false);
        }
    }

    // epilogue: bias + optional SiLU, scatter per ISA C/D layout
#pragma unroll
    for (int mt = 0; mt < 2; ++mt) {
#pragma unroll
        for (int nt = 0; nt < 4; ++nt) {
            int col = bn + wn + nt * 16 + lrow;
            float bv = bias[col];
#pragma unroll
            for (int r = 0; r < 8; ++r) {
                int row = bm + wm + mt * 16 + lhalf * 8 + r;
                float v = acc[mt][nt][r] + bv;
                if (act == 2) v = v / (1.f + __expf(-v));
                C[(size_t)row * Ndim + col] = v;
            }
        }
    }
}

// ---------------------------------------------------------------------------
// sioconv selective recurrence: h_t = a_t h_{t-1} + (1-a_t) z_t, a=sigmoid(apre)
// ---------------------------------------------------------------------------
__global__ __launch_bounds__(256) void sioconv_scan_kernel(
    const float* __restrict__ z, const float* __restrict__ apre,
    float* __restrict__ h)
{
    int idx = blockIdx.x * 256 + threadIdx.x;     // 0..4095
    int b = idx >> 10;
    int d = idx & (DIM - 1);
    size_t base = (size_t)b * SEQ * DIM + d;
    float hs = 0.f;
    for (int s = 0; s < SEQ; ++s) {
        size_t off = base + (size_t)s * DIM;
        float a = 1.f / (1.f + __expf(-apre[off]));
        hs = a * hs + (1.f - a) * z[off];
        h[off] = hs;
    }
}

// ---------------------------------------------------------------------------
// normalize k rows of HDIM per head:  k[seg,:] /= (||k[seg,:]|| + eps)
// ---------------------------------------------------------------------------
__global__ __launch_bounds__(256) void knorm_kernel(float* __restrict__ k)
{
    int seg  = blockIdx.x * 8 + (threadIdx.x >> 5);   // MTOK*NHEAD segments
    int lane = threadIdx.x & 31;
    float* p = k + (size_t)seg * HDIM;
    float s = 0.f;
    for (int i = lane; i < HDIM; i += 32) { float v = p[i]; s += v * v; }
    for (int o = 16; o > 0; o >>= 1) s += __shfl_xor(s, o, 32);
    float inv = 1.f / (sqrtf(s) + 1e-6f);
    for (int i = lane; i < HDIM; i += 32) p[i] *= inv;
}

// ---------------------------------------------------------------------------
// lr / wd projections (N=8, small):  thread per (token, head)
// ---------------------------------------------------------------------------
__global__ __launch_bounds__(256) void lrwd_kernel(
    const bf16_t* __restrict__ xn,
    const float* __restrict__ wlr, const float* __restrict__ blr,
    const float* __restrict__ wwd, const float* __restrict__ bwd,
    float* __restrict__ lr, float* __restrict__ wd)
{
    int idx = blockIdx.x * 256 + threadIdx.x;     // MTOK*NHEAD
    int t = idx >> 3;
    int h = idx & 7;
    const bf16_t* xr = xn + (size_t)t * DIM;
    float s1 = 0.f, s2 = 0.f;
    for (int k = 0; k < DIM; ++k) {
        float v = (float)xr[k];
        s1 += v * wlr[k * NHEAD + h];
        s2 += v * wwd[k * NHEAD + h];
    }
    lr[idx] = 0.5f  / (1.f + __expf(-(s1 + blr[h])));
    wd[idx] = 0.01f / (1.f + __expf(-(s2 + bwd[h])));
}

// ---------------------------------------------------------------------------
// chunkwise delta-rule neural memory: one block per (batch, head),
// M-state + chunk tiles resident in dynamic LDS (~181 KB),
// chunk tiles staged with async global->LDS copies.
// ---------------------------------------------------------------------------
__global__ __launch_bounds__(256) void neural_memory_kernel(
    const float* __restrict__ kbuf, const float* __restrict__ vbuf,
    const float* __restrict__ qbuf, const float* __restrict__ lrbuf,
    const float* __restrict__ wdbuf, float* __restrict__ ybuf)
{
    extern __shared__ float smem[];
    float* Ms  = smem;              // 128*128
    float* ks  = Ms  + 16384;       // 64*128
    float* vs  = ks  + 8192;        // 64*128 (err in place)
    float* qs  = vs  + 8192;        // 64*128
    float* wsm = qs  + 8192;        // 64*64
    float* lrs = wsm + 4096;        // 64
    float* wds = lrs + 64;          // 64 (later reused for scaled-lr)
    float* Gs  = wds + 64;          // 64

    const int tid = threadIdx.x;
    const int bh  = blockIdx.x;     // 0..31
    const int b   = bh >> 3;
    const int h   = bh & 7;

    for (int i = tid; i < 16384; i += 256) Ms[i] = 0.f;
    __syncthreads();

    for (int c = 0; c < NC; ++c) {
        const int t0 = c * CHUNK;
        // async-stage k/v/q chunk tiles: 64 rows x 128 f32 = 2048 x 16B segs each
        for (int i = tid; i < 2048; i += 256) {
            int t = i >> 5, cc = i & 31;
            size_t g = (size_t)(b * SEQ + t0 + t) * DIM + h * HDIM + cc * 4;
            async_load_b128(kbuf + g, ks + t * HDIM + cc * 4);
            async_load_b128(vbuf + g, vs + t * HDIM + cc * 4);
            async_load_b128(qbuf + g, qs + t * HDIM + cc * 4);
        }
        if (tid < CHUNK) {
            int gi = (b * SEQ + t0 + tid) * NHEAD + h;
            lrs[tid] = lrbuf[gi];
            wds[tid] = wdbuf[gi];
        }
        wait_async0();
        __syncthreads();
        if (tid == 0) {
            float gacc = 1.f;
            for (int t = 0; t < CHUNK; ++t) { gacc *= (1.f - wds[t]); Gs[t] = gacc; }
        }
        __syncthreads();

        // err = v - k @ M   (in place into vs)
        for (int i = tid; i < 8192; i += 256) {
            int t = i >> 7, e = i & 127;
            float s = 0.f;
            for (int d = 0; d < HDIM; ++d) s += ks[t * HDIM + d] * Ms[d * HDIM + e];
            vs[i] -= s;
        }
        __syncthreads();

        // w[t,s] = (q_t.k_s) * tril * (G_t/G_s) * lr_s
        for (int i = tid; i < 4096; i += 256) {
            int t = i >> 6, ss = i & 63;
            float w = 0.f;
            if (ss <= t) {
                float d = 0.f;
                for (int e = 0; e < HDIM; ++e) d += qs[t * HDIM + e] * ks[ss * HDIM + e];
                w = d * (Gs[t] / Gs[ss]) * lrs[ss];
            }
            wsm[i] = w;
        }
        __syncthreads();

        // y = G_t * q @ M + w @ err
        for (int i = tid; i < 8192; i += 256) {
            int t = i >> 7, e = i & 127;
            float d = 0.f;
            for (int dd = 0; dd < HDIM; ++dd) d += qs[t * HDIM + dd] * Ms[dd * HDIM + e];
            float acc = Gs[t] * d;
            for (int ss = 0; ss < CHUNK; ++ss) acc += wsm[t * 64 + ss] * vs[ss * HDIM + e];
            ybuf[(size_t)(b * SEQ + t0 + t) * DIM + h * HDIM + e] = acc;
        }
        __syncthreads();

        // scaled lr for the state update
        float Gend = Gs[CHUNK - 1];
        if (tid < CHUNK) wds[tid] = (Gend / Gs[tid]) * lrs[tid];
        __syncthreads();

        // M <- Gend*M + k^T @ (slr * err)
        for (int i = tid; i < 16384; i += 256) {
            int d = i >> 7, e = i & 127;
            float acc = Gend * Ms[i];
            for (int ss = 0; ss < CHUNK; ++ss)
                acc += wds[ss] * ks[ss * HDIM + d] * vs[ss * HDIM + e];
            Ms[i] = acc;
        }
        __syncthreads();
    }
}

// ---------------------------------------------------------------------------
// elementwise helpers
// ---------------------------------------------------------------------------
__global__ void f32_to_bf16_kernel(const float* __restrict__ in,
                                   bf16_t* __restrict__ out, int n)
{
    int i = blockIdx.x * 256 + threadIdx.x;
    if (i < n) out[i] = (bf16_t)in[i];
}

// out = x + a * g   (g already silu'd in GEMM epilogue)
__global__ void combine_gate_kernel(const float* __restrict__ x,
                                    const float* __restrict__ a,
                                    const float* __restrict__ g,
                                    float* __restrict__ out, int n)
{
    int i = blockIdx.x * 256 + threadIdx.x;
    if (i < n) out[i] = x[i] + a[i] * g[i];
}

__global__ void add_kernel(const float* __restrict__ a,
                           const float* __restrict__ b,
                           float* __restrict__ out, int n)
{
    int i = blockIdx.x * 256 + threadIdx.x;
    if (i < n) out[i] = a[i] + b[i];
}

// out_bf16 = u * g  (g already silu'd)
__global__ void mul_to_bf16_kernel(const float* __restrict__ u,
                                   const float* __restrict__ g,
                                   bf16_t* __restrict__ out, int n)
{
    int i = blockIdx.x * 256 + threadIdx.x;
    if (i < n) out[i] = (bf16_t)(u[i] * g[i]);
}

// ---------------------------------------------------------------------------
// host side
// ---------------------------------------------------------------------------
extern "C" void kernel_launch(void* const* d_in, const int* in_sizes, int n_in,
                              void* d_out, int out_size, void* d_ws, size_t ws_size,
                              hipStream_t stream)
{
    (void)in_sizes; (void)n_in; (void)out_size; (void)ws_size;

    const float* x    = (const float*)d_in[0];
    const float* wz   = (const float*)d_in[1];  const float* bz  = (const float*)d_in[2];
    const float* wa   = (const float*)d_in[3];  const float* ba  = (const float*)d_in[4];
    const float* wy   = (const float*)d_in[5];  const float* by  = (const float*)d_in[6];
    const float* wg   = (const float*)d_in[7];  const float* bg  = (const float*)d_in[8];
    const float* wk   = (const float*)d_in[9];  const float* bk  = (const float*)d_in[10];
    const float* wv   = (const float*)d_in[11]; const float* bv  = (const float*)d_in[12];
    const float* wq   = (const float*)d_in[13]; const float* bq  = (const float*)d_in[14];
    const float* wlr  = (const float*)d_in[15]; const float* blr = (const float*)d_in[16];
    const float* wwd  = (const float*)d_in[17]; const float* bwd = (const float*)d_in[18];
    const float* wo   = (const float*)d_in[19]; const float* bo  = (const float*)d_in[20];
    const float* w1   = (const float*)d_in[21]; const float* b1  = (const float*)d_in[22];
    const float* wg1  = (const float*)d_in[23]; const float* bg1 = (const float*)d_in[24];
    const float* w2   = (const float*)d_in[25]; const float* b2  = (const float*)d_in[26];
    const float* gsio = (const float*)d_in[27];
    const float* gmem = (const float*)d_in[28];
    const float* gffn = (const float*)d_in[29];
    float* out = (float*)d_out;

    const size_t MD = (size_t)MTOK * DIM;   // 8,388,608
    const size_t MF = (size_t)MTOK * FFH;   // 33,554,432

    char* ws = (char*)d_ws;
    size_t off = 0;
    auto alloc = [&](size_t bytes) { char* p = ws + off; off += (bytes + 255) & ~(size_t)255; return p; };

    float*  h1    = (float*)alloc(MD * 4);
    float*  h2    = (float*)alloc(MD * 4);
    float*  tA    = (float*)alloc(MD * 4);
    float*  tB    = (float*)alloc(MD * 4);
    float*  tC    = (float*)alloc(MD * 4);
    float*  tD    = (float*)alloc(MD * 4);
    float*  ubuf  = (float*)alloc(MF * 4);
    float*  gbuf  = (float*)alloc(MF * 4);
    bf16_t* abf   = (bf16_t*)alloc(MF * 2);
    bf16_t* wtz   = (bf16_t*)alloc((size_t)DIM * DIM * 2);
    bf16_t* wta   = (bf16_t*)alloc((size_t)DIM * DIM * 2);
    bf16_t* wty   = (bf16_t*)alloc((size_t)DIM * DIM * 2);
    bf16_t* wtg   = (bf16_t*)alloc((size_t)DIM * DIM * 2);
    bf16_t* wtk   = (bf16_t*)alloc((size_t)DIM * DIM * 2);
    bf16_t* wtv   = (bf16_t*)alloc((size_t)DIM * DIM * 2);
    bf16_t* wtq   = (bf16_t*)alloc((size_t)DIM * DIM * 2);
    bf16_t* wto   = (bf16_t*)alloc((size_t)DIM * DIM * 2);
    bf16_t* wt1   = (bf16_t*)alloc((size_t)FFH * DIM * 2);
    bf16_t* wtg1  = (bf16_t*)alloc((size_t)FFH * DIM * 2);
    bf16_t* wt2   = (bf16_t*)alloc((size_t)DIM * FFH * 2);
    float*  lrb   = (float*)alloc((size_t)MTOK * NHEAD * 4);
    float*  wdb   = (float*)alloc((size_t)MTOK * NHEAD * 4);

    dim3 tpT(32, 8);
    dim3 grDD(DIM / 32, DIM / 32);
    hipLaunchKernelGGL(transpose_to_bf16_kernel, grDD, tpT, 0, stream, wz, wtz, DIM, DIM);
    hipLaunchKernelGGL(transpose_to_bf16_kernel, grDD, tpT, 0, stream, wa, wta, DIM, DIM);
    hipLaunchKernelGGL(transpose_to_bf16_kernel, grDD, tpT, 0, stream, wy, wty, DIM, DIM);
    hipLaunchKernelGGL(transpose_to_bf16_kernel, grDD, tpT, 0, stream, wg, wtg, DIM, DIM);
    hipLaunchKernelGGL(transpose_to_bf16_kernel, grDD, tpT, 0, stream, wk, wtk, DIM, DIM);
    hipLaunchKernelGGL(transpose_to_bf16_kernel, grDD, tpT, 0, stream, wv, wtv, DIM, DIM);
    hipLaunchKernelGGL(transpose_to_bf16_kernel, grDD, tpT, 0, stream, wq, wtq, DIM, DIM);
    hipLaunchKernelGGL(transpose_to_bf16_kernel, grDD, tpT, 0, stream, wo, wto, DIM, DIM);
    hipLaunchKernelGGL(transpose_to_bf16_kernel, dim3(FFH / 32, DIM / 32), tpT, 0, stream, w1,  wt1,  DIM, FFH);
    hipLaunchKernelGGL(transpose_to_bf16_kernel, dim3(FFH / 32, DIM / 32), tpT, 0, stream, wg1, wtg1, DIM, FFH);
    hipLaunchKernelGGL(transpose_to_bf16_kernel, dim3(DIM / 32, FFH / 32), tpT, 0, stream, w2,  wt2,  FFH, DIM);

    const dim3 gemmDD(DIM / 128, MTOK / 128);   // (8, 64)
    const dim3 gemmDF(FFH / 128, MTOK / 128);   // (32, 64)
    const int nMD = (int)MD, nMF = (int)MF;
    const int ewMD = (nMD + 255) / 256, ewMF = (nMF + 255) / 256;

    // ---- stage 1: sioconv -------------------------------------------------
    hipLaunchKernelGGL(rmsnorm_bf16_kernel, dim3(MTOK), dim3(256), 0, stream, x, gsio, abf);
    hipLaunchKernelGGL(gemm_bf16_wmma_kernel, gemmDD, dim3(256), 0, stream, abf, wtz, bz, tA, MTOK, DIM, DIM, 0);
    hipLaunchKernelGGL(gemm_bf16_wmma_kernel, gemmDD, dim3(256), 0, stream, abf, wta, ba, tB, MTOK, DIM, DIM, 0);
    hipLaunchKernelGGL(gemm_bf16_wmma_kernel, gemmDD, dim3(256), 0, stream, abf, wtg, bg, tC, MTOK, DIM, DIM, 2);
    hipLaunchKernelGGL(sioconv_scan_kernel, dim3(BATCH * DIM / 256), dim3(256), 0, stream, tA, tB, tD);
    hipLaunchKernelGGL(f32_to_bf16_kernel, dim3(ewMD), dim3(256), 0, stream, tD, abf, nMD);
    hipLaunchKernelGGL(gemm_bf16_wmma_kernel, gemmDD, dim3(256), 0, stream, abf, wty, by, tA, MTOK, DIM, DIM, 0);
    hipLaunchKernelGGL(combine_gate_kernel, dim3(ewMD), dim3(256), 0, stream, x, tA, tC, h1, nMD);

    // ---- stage 2: neural memory ------------------------------------------
    hipLaunchKernelGGL(rmsnorm_bf16_kernel, dim3(MTOK), dim3(256), 0, stream, h1, gmem, abf);
    hipLaunchKernelGGL(gemm_bf16_wmma_kernel, gemmDD, dim3(256), 0, stream, abf, wtk, bk, tB, MTOK, DIM, DIM, 0);
    hipLaunchKernelGGL(gemm_bf16_wmma_kernel, gemmDD, dim3(256), 0, stream, abf, wtv, bv, tC, MTOK, DIM, DIM, 0);
    hipLaunchKernelGGL(gemm_bf16_wmma_kernel, gemmDD, dim3(256), 0, stream, abf, wtq, bq, tD, MTOK, DIM, DIM, 0);
    hipLaunchKernelGGL(knorm_kernel, dim3(MTOK * NHEAD / 8), dim3(256), 0, stream, tB);
    hipLaunchKernelGGL(lrwd_kernel, dim3(MTOK * NHEAD / 256), dim3(256), 0, stream,
                       abf, wlr, blr, wwd, bwd, lrb, wdb);
    const size_t memLds = (size_t)(16384 + 3 * 8192 + 4096 + 3 * 64) * sizeof(float);
    hipLaunchKernelGGL(neural_memory_kernel, dim3(BATCH * NHEAD), dim3(256), memLds, stream,
                       tB, tC, tD, lrb, wdb, tA);
    hipLaunchKernelGGL(f32_to_bf16_kernel, dim3(ewMD), dim3(256), 0, stream, tA, abf, nMD);
    hipLaunchKernelGGL(gemm_bf16_wmma_kernel, gemmDD, dim3(256), 0, stream, abf, wto, bo, tB, MTOK, DIM, DIM, 0);
    hipLaunchKernelGGL(add_kernel, dim3(ewMD), dim3(256), 0, stream, h1, tB, h2, nMD);

    // ---- stage 3: ffn -----------------------------------------------------
    hipLaunchKernelGGL(rmsnorm_bf16_kernel, dim3(MTOK), dim3(256), 0, stream, h2, gffn, abf);
    hipLaunchKernelGGL(gemm_bf16_wmma_kernel, gemmDF, dim3(256), 0, stream, abf, wt1,  b1,  ubuf, MTOK, FFH, DIM, 0);
    hipLaunchKernelGGL(gemm_bf16_wmma_kernel, gemmDF, dim3(256), 0, stream, abf, wtg1, bg1, gbuf, MTOK, FFH, DIM, 2);
    hipLaunchKernelGGL(mul_to_bf16_kernel, dim3(ewMF), dim3(256), 0, stream, ubuf, gbuf, abf, nMF);
    hipLaunchKernelGGL(gemm_bf16_wmma_kernel, gemmDD, dim3(256), 0, stream, abf, wt2, b2, tA, MTOK, DIM, FFH, 0);
    hipLaunchKernelGGL(add_kernel, dim3(ewMD), dim3(256), 0, stream, h2, tA, out, nMD);
}
